// AttentionHead_32908039422116
// MI455X (gfx1250) — matrix operand order repaired and checked by reading, not compiled, not run on hardware
//
#include <hip/hip_runtime.h>
#include <math.h>

typedef __attribute__((ext_vector_type(16))) __bf16 v16bf;
typedef __attribute__((ext_vector_type(8)))  __bf16 v8bf;
typedef __attribute__((ext_vector_type(4)))  __bf16 v4bf;
typedef __attribute__((ext_vector_type(8)))  float  v8f;
typedef __attribute__((ext_vector_type(4)))  unsigned int v4u;
typedef __attribute__((ext_vector_type(8)))  int    v8i;
typedef __attribute__((ext_vector_type(4)))  int    v4i;

#define SEQ 4096
#define DM  2048
#define DH  128

#if defined(__has_include)
#if __has_include(<hip/amd_detail/amd_gfx1250_TDM.h>)
#define TDM_6ARG 1
#endif
#endif

__device__ __forceinline__ v16bf cat8(v8bf a, v8bf b) {
  return __builtin_shufflevector(a, b, 0,1,2,3,4,5,6,7,8,9,10,11,12,13,14,15);
}

__device__ __forceinline__ v8bf cvt8(float4 a, float4 b) {
  v8bf r;
  r[0] = (__bf16)a.x; r[1] = (__bf16)a.y; r[2] = (__bf16)a.z; r[3] = (__bf16)a.w;
  r[4] = (__bf16)b.x; r[5] = (__bf16)b.y; r[6] = (__bf16)b.z; r[7] = (__bf16)b.w;
  return r;
}

// ---------------------------------------------------------------------------
// TDM: async 2D tile load (bf16 elements) from global to LDS.
// D# layout per CDNA5 ISA.
//   group0: count=1 | lds_addr | global_addr[56:0] | type=2
//   group1: data_size=2B, tensor_dim0/1, tile_dim0/1, tensor_dim0_stride
// ---------------------------------------------------------------------------
__device__ __forceinline__ void tdm_load_2d_bf16(
    unsigned lds_off, const void* gaddr,
    unsigned tensor_d0, unsigned tensor_d1, unsigned stride0,
    unsigned tile_d0, unsigned tile_d1)
{
  const unsigned long long ga = (unsigned long long)gaddr;
  v4u g0;
  g0[0] = 1u;                                          // count=1, user mode
  g0[1] = lds_off;                                     // lds_addr (bytes)
  g0[2] = (unsigned)(ga & 0xFFFFFFFFu);                // global_addr[31:0]
  g0[3] = (unsigned)((ga >> 32) & 0x1FFFFFFu) | (2u << 30);  // [56:32] | type=2
  v8i g1;
  g1[0] = (int)(1u << 16);                             // wg_mask=0, data_size=1 (2B)
  g1[1] = (int)(tensor_d0 << 16);                      // tensor_dim0[15:0] @bit48
  g1[2] = (int)((tensor_d0 >> 16) | (tensor_d1 << 16));// dim0[31:16] | dim1[15:0]
  g1[3] = (int)((tensor_d1 >> 16) | (tile_d0 << 16));  // dim1[31:16] | tile_dim0
  g1[4] = (int)(tile_d1 & 0xFFFFu);                    // tile_dim1, tile_dim2=0
  g1[5] = (int)stride0;                                // tensor_dim0_stride[31:0]
  g1[6] = 0;
  g1[7] = 0;
  v4i z = {0, 0, 0, 0};
#if defined(TDM_6ARG)
  v8i z8 = {0, 0, 0, 0, 0, 0, 0, 0};
  __builtin_amdgcn_tensor_load_to_lds(g0, g1, z, z, z8, 0);
#else
  __builtin_amdgcn_tensor_load_to_lds(g0, g1, z, z, 0);
#endif
}

// ---------------------------------------------------------------------------
// One-shot: convert f32 weight matrices to bf16 (removes per-wave re-convert
// and halves L2 weight traffic in the projection GEMMs).
// ---------------------------------------------------------------------------
__global__ __launch_bounds__(256) void cvt_w_bf16(const float* __restrict__ in,
                                                  __bf16* __restrict__ out)
{
  const int i = (blockIdx.x * 256 + threadIdx.x) * 4;   // n = 128*2048
  const float4 v = *(const float4*)(in + i);
  v4bf o;
  o[0] = (__bf16)v.x; o[1] = (__bf16)v.y; o[2] = (__bf16)v.z; o[3] = (__bf16)v.w;
  *(v4bf*)(out + i) = o;
}

// ---------------------------------------------------------------------------
// Projection: out = X[4096,2048] @ W[128,2048]^T + bias (bf16 weights).
// One wave = 16 rows x 128 cols; A fragment reused across 8 N-tiles.
// transpose_out=1 stores out[n*SEQ+m] (for V), else out[m*DH+n].
// ---------------------------------------------------------------------------
__global__ __launch_bounds__(128) void proj_bf16_wmma(
    const float* __restrict__ X, const __bf16* __restrict__ Wb,
    const float* __restrict__ bias, __bf16* __restrict__ out,
    int transpose_out, float scale)
{
  const int lane = threadIdx.x & 31;
  const int wave = threadIdx.x >> 5;
  const int hl   = lane >> 4;
  const int l16  = lane & 15;
  const int m0   = (blockIdx.x * 4 + wave) * 16;       // 256 row tiles

  v8f acc[8];
#pragma unroll
  for (int nt = 0; nt < 8; ++nt) {
    const float bv = bias[nt * 16 + l16];
#pragma unroll
    for (int v = 0; v < 8; ++v) acc[nt][v] = bv;
  }

  const float* xrow = X + (size_t)(m0 + l16) * DM;

  for (int ks = 0; ks < DM / 32; ++ks) {
    const int kbase = ks * 32;
    // A-layout (16x32): lane holds K = hl*16 + {0..15}
    const float4* p0 = (const float4*)(xrow + kbase + hl * 16);
    const float4* p1 = (const float4*)(xrow + kbase + hl * 16 + 8);
    const v16bf a = cat8(cvt8(p0[0], p0[1]), cvt8(p1[0], p1[1]));

#pragma unroll
    for (int nt = 0; nt < 8; ++nt) {
      // B-layout (32x16): lane holds col n = l16, K = hl*16 + {0..15}
      const v16bf bf = *(const v16bf*)(Wb + (size_t)(nt * 16 + l16) * DM +
                                       kbase + hl * 16);
      acc[nt] = __builtin_amdgcn_wmma_f32_16x16x32_bf16(
          false, a, false, bf, (short)0, acc[nt], false, false);
    }
  }

#pragma unroll
  for (int nt = 0; nt < 8; ++nt) {
#pragma unroll
    for (int v = 0; v < 8; ++v) {
      const int m = m0 + v + 8 * hl;
      const int n = nt * 16 + l16;
      const __bf16 val = (__bf16)(acc[nt][v] * scale);
      if (transpose_out) out[(size_t)n * SEQ + m] = val;
      else               out[(size_t)m * DH  + n] = val;
    }
  }
}

// ---------------------------------------------------------------------------
// Flash attention forward, causal. One wave = 16 query rows; key blocks of 32.
// K and V tiles are DMA'd into LDS by the Tensor Data Mover, double-buffered
// (issue next block's loads, s_wait_tensorcnt 2 gates the current buffer).
// P is staged through LDS to convert score (C) layout -> WMMA A layout.
// ---------------------------------------------------------------------------
__global__ __launch_bounds__(64) void attn_fwd_wmma(
    const __bf16* __restrict__ Q, const __bf16* __restrict__ K,
    const __bf16* __restrict__ Vt, float* __restrict__ out)
{
  __shared__ __bf16 kbuf[2][2][32 * DH];    // [wave][buf] 32x128 tile (8KB)
  __shared__ __bf16 vbuf[2][2][DH * 32];    // [wave][buf] 128x32 tile (8KB)
  __shared__ __bf16 pstage[2][16 * 32];

  const int lane = threadIdx.x & 31;
  const int wave = threadIdx.x >> 5;
  const int hl   = lane >> 4;
  const int l16  = lane & 15;
  const int m0   = (blockIdx.x * 2 + wave) * 16;

  // Resident Q fragments (16 x 128 bf16 as 4 A-fragments of 16x32)
  v16bf qf[4];
  const __bf16* qrow = Q + (size_t)(m0 + l16) * DH;
#pragma unroll
  for (int f = 0; f < 4; ++f) {
    const v8bf c0 = *(const v8bf*)(qrow + f * 32 + hl * 16);
    const v8bf c1 = *(const v8bf*)(qrow + f * 32 + hl * 16 + 8);
    qf[f] = cat8(c0, c1);
  }

  v8f acc[8];
#pragma unroll
  for (int nt = 0; nt < 8; ++nt)
#pragma unroll
    for (int v = 0; v < 8; ++v) acc[nt][v] = 0.0f;

  float Mrow[8], Lrow[8];
#pragma unroll
  for (int v = 0; v < 8; ++v) { Mrow[v] = -1e30f; Lrow[v] = 0.0f; }

  const int nblocks = ((m0 + 15) >> 5) + 1;   // causal key-block count

  // Prime buffer 0: K tile 32x128 (row-major), Vt tile 128x32 (stride SEQ)
  tdm_load_2d_bf16((unsigned)(size_t)&kbuf[wave][0][0], K + (size_t)0 * DH,
                   DH, SEQ, DH, DH, 32);
  tdm_load_2d_bf16((unsigned)(size_t)&vbuf[wave][0][0], Vt + 0,
                   SEQ, DH, SEQ, 32, DH);

  for (int j = 0; j < nblocks; ++j) {
    const int j0 = j * 32;
    if (j + 1 < nblocks) {
      const int n0 = j0 + 32;
      tdm_load_2d_bf16((unsigned)(size_t)&kbuf[wave][(j + 1) & 1][0],
                       K + (size_t)n0 * DH, DH, SEQ - n0, DH, DH, 32);
      tdm_load_2d_bf16((unsigned)(size_t)&vbuf[wave][(j + 1) & 1][0],
                       Vt + n0, SEQ - n0, DH, SEQ, 32, DH);
      __builtin_amdgcn_s_wait_tensorcnt(2);   // current buffer complete
    } else {
      __builtin_amdgcn_s_wait_tensorcnt(0);
    }
    const __bf16* kt = &kbuf[wave][j & 1][0];
    const __bf16* vt = &vbuf[wave][j & 1][0];

    // ---- S = Q @ K^T (two 16x16 tiles covering 32 keys) ----
    v8f s[2];
#pragma unroll
    for (int t = 0; t < 2; ++t) {
      v8f sa;
#pragma unroll
      for (int v = 0; v < 8; ++v) sa[v] = 0.0f;
#pragma unroll
      for (int f = 0; f < 4; ++f) {
        const v16bf bf = *(const v16bf*)(kt + (t * 16 + l16) * DH +
                                         f * 32 + hl * 16);
        sa = __builtin_amdgcn_wmma_f32_16x16x32_bf16(
            false, qf[f], false, bf, (short)0, sa, false, false);
      }
      const int n_g = j0 + t * 16 + l16;
#pragma unroll
      for (int v = 0; v < 8; ++v) {
        const int m_g = m0 + v + 8 * hl;
        if (n_g > m_g) sa[v] = -1e30f;          // causal mask
      }
      s[t] = sa;
    }

    // ---- online softmax (row stats per VGPR-index, 16-lane reductions) ----
    float scalef[8];
#pragma unroll
    for (int v = 0; v < 8; ++v) {
      float mx = fmaxf(s[0][v], s[1][v]);
      mx = fmaxf(mx, __shfl_xor(mx, 1, 32));
      mx = fmaxf(mx, __shfl_xor(mx, 2, 32));
      mx = fmaxf(mx, __shfl_xor(mx, 4, 32));
      mx = fmaxf(mx, __shfl_xor(mx, 8, 32));
      const float newM = fmaxf(Mrow[v], mx);
      scalef[v] = __expf(Mrow[v] - newM);
      Mrow[v] = newM;
      const float p0 = __expf(s[0][v] - newM);
      const float p1 = __expf(s[1][v] - newM);
      s[0][v] = p0; s[1][v] = p1;
      float rs = p0 + p1;
      rs += __shfl_xor(rs, 1, 32);
      rs += __shfl_xor(rs, 2, 32);
      rs += __shfl_xor(rs, 4, 32);
      rs += __shfl_xor(rs, 8, 32);
      Lrow[v] = Lrow[v] * scalef[v] + rs;
    }
#pragma unroll
    for (int nt = 0; nt < 8; ++nt)
#pragma unroll
      for (int v = 0; v < 8; ++v) acc[nt][v] *= scalef[v];

    // ---- stage P (16x32) through LDS: C layout -> A layout bf16 ----
    __bf16* pw = &pstage[wave][0];
#pragma unroll
    for (int t = 0; t < 2; ++t)
#pragma unroll
      for (int v = 0; v < 8; ++v)
        pw[(v + 8 * hl) * 32 + t * 16 + l16] = (__bf16)s[t][v];
    asm volatile("s_wait_dscnt 0" ::: "memory");
    v16bf pa;
    {
      const __bf16* pr = pw + l16 * 32 + hl * 16;
      pa = cat8(*(const v8bf*)(pr), *(const v8bf*)(pr + 8));
    }

    // ---- O += P @ V (V tile in LDS, contiguous along keys) ----
#pragma unroll
    for (int nt = 0; nt < 8; ++nt) {
      const v16bf vf = *(const v16bf*)(vt + (nt * 16 + l16) * 32 + hl * 16);
      acc[nt] = __builtin_amdgcn_wmma_f32_16x16x32_bf16(
          false, pa, false, vf, (short)0, acc[nt], false, false);
    }
  }

  // ---- epilogue: divide by row sum, store f32 ----
#pragma unroll
  for (int nt = 0; nt < 8; ++nt)
#pragma unroll
    for (int v = 0; v < 8; ++v) {
      const int m = m0 + v + 8 * hl;
      const int n = nt * 16 + l16;
      out[(size_t)m * DH + n] = acc[nt][v] / Lrow[v];
    }
}

// ---------------------------------------------------------------------------
extern "C" void kernel_launch(void* const* d_in, const int* in_sizes, int n_in,
                              void* d_out, int out_size, void* d_ws, size_t ws_size,
                              hipStream_t stream) {
  (void)in_sizes; (void)n_in; (void)out_size; (void)ws_size;
  const float* in_q = (const float*)d_in[0];
  const float* in_k = (const float*)d_in[1];
  const float* in_v = (const float*)d_in[2];
  const float* WQw  = (const float*)d_in[3];
  const float* WQb  = (const float*)d_in[4];
  const float* WKw  = (const float*)d_in[5];
  const float* WKb  = (const float*)d_in[6];
  const float* WVw  = (const float*)d_in[7];
  const float* WVb  = (const float*)d_in[8];

  char* ws = (char*)d_ws;
  __bf16* Qbf  = (__bf16*)(ws);                            // [4096,128] 1MB
  __bf16* Kbf  = (__bf16*)(ws + (size_t)(1 << 20));        // [4096,128] 1MB
  __bf16* Vt   = (__bf16*)(ws + (size_t)(2 << 20));        // [128,4096] 1MB
  __bf16* WqB  = (__bf16*)(ws + (size_t)(3 << 20));        // 512KB
  __bf16* WkB  = (__bf16*)(ws + (size_t)(3 << 20) + (512 << 10));
  __bf16* WvB  = (__bf16*)(ws + (size_t)(4 << 20));

  const float qscale = 1.0f / sqrtf((float)DM);  // fold /sqrt(d_model) into Q

  cvt_w_bf16<<<256, 256, 0, stream>>>(WQw, WqB);   // 128*2048 / (256*4)
  cvt_w_bf16<<<256, 256, 0, stream>>>(WKw, WkB);
  cvt_w_bf16<<<256, 256, 0, stream>>>(WVw, WvB);

  proj_bf16_wmma<<<64, 128, 0, stream>>>(in_q, WqB, WQb, Qbf, 0, qscale);
  proj_bf16_wmma<<<64, 128, 0, stream>>>(in_k, WkB, WKb, Kbf, 0, 1.0f);
  proj_bf16_wmma<<<64, 128, 0, stream>>>(in_v, WvB, WVb, Vt,  1, 1.0f);

  attn_fwd_wmma<<<128, 64, 0, stream>>>(Qbf, Kbf, Vt, (float*)d_out);
}
